// GraphModule_57767310131620
// MI455X (gfx1250) — matrix operand order repaired and checked by reading, not compiled, not run on hardware
//
#include <hip/hip_runtime.h>
#include <hip/hip_bf16.h>

#define IN_DIM  256
#define OUT_DIM 256
#define BB      8
#define NNODE   1024
#define NE      262144
#define NTOT    (BB * NNODE)   /* 8192 */
#define KTOP    512
#define SELU_SCALE 1.0507009873554805f
#define SELU_ALPHA 1.6732632423543772f

typedef __bf16 bf16_t;
typedef bf16_t v16bf __attribute__((ext_vector_type(16)));
typedef float  v8f   __attribute__((ext_vector_type(8)));

__device__ __forceinline__ unsigned ordU(float f) {
    unsigned u = __float_as_uint(f);
    return (u & 0x80000000u) ? ~u : (u | 0x80000000u);
}
__device__ __forceinline__ float unordU(unsigned u) {
    unsigned b = (u & 0x80000000u) ? (u ^ 0x80000000u) : ~u;
    return __uint_as_float(b);
}

// ---------------------------------------------------------------------------
// Kernel 1: per-edge alpha = att_map . tanh(W (x_i*x_j) + b), bf16 WMMA.
// One wave = 32 edges (TWO 16-row M tiles) so each LDS B-fragment feeds two
// WMMAs; 16 N-tiles x 8 K-tiles. sched_group_barrier pipelines DS reads two
// K-steps ahead of the matrix ops.
// ---------------------------------------------------------------------------
__global__ __launch_bounds__(256) void k_edge_alpha(
    const float* __restrict__ x, const int* __restrict__ src,
    const int* __restrict__ dst, const float* __restrict__ Wg,
    const float* __restrict__ att_b, const float* __restrict__ att_map,
    float* __restrict__ alpha)
{
    __shared__ __align__(32) bf16_t Wl[OUT_DIM * IN_DIM];   // 128 KB
    __shared__ float abL[OUT_DIM];
    __shared__ float amL[OUT_DIM];
    for (int i = threadIdx.x; i < OUT_DIM * IN_DIM; i += 256)
        Wl[i] = (bf16_t)Wg[i];
    if (threadIdx.x < OUT_DIM) {
        abL[threadIdx.x] = att_b[threadIdx.x];
        amL[threadIdx.x] = att_map[threadIdx.x];
    }
    __syncthreads();

    const int wave  = threadIdx.x >> 5;
    const int lane  = threadIdx.x & 31;
    const int group = blockIdx.x * 8 + wave;   // 32 edges per group
    const int row   = lane & 15;
    const int hi    = lane >> 4;

    const int e0 = group * 32 + row;
    const int e1 = e0 + 16;
    const float* __restrict__ xs0 = x + (size_t)src[e0] * IN_DIM;
    const float* __restrict__ xd0 = x + (size_t)dst[e0] * IN_DIM;
    const float* __restrict__ xs1 = x + (size_t)src[e1] * IN_DIM;
    const float* __restrict__ xd1 = x + (size_t)dst[e1] * IN_DIM;

    // A fragments (product x_i * x_j -> bf16) for both M tiles, all 8 K-tiles
    v16bf a0[8], a1[8];
#pragma unroll
    for (int kt = 0; kt < 8; ++kt) {
        const int kb0 = kt * 32 + hi * 16;
        const int kb1 = kb0 + 8;
#pragma unroll
        for (int j = 0; j < 8; ++j) {
            a0[kt][j]     = (bf16_t)(xs0[kb0 + j] * xd0[kb0 + j]);
            a0[kt][8 + j] = (bf16_t)(xs0[kb1 + j] * xd0[kb1 + j]);
            a1[kt][j]     = (bf16_t)(xs1[kb0 + j] * xd1[kb0 + j]);
            a1[kt][8 + j] = (bf16_t)(xs1[kb1 + j] * xd1[kb1 + j]);
        }
    }

    float pal0[8] = {0.f, 0.f, 0.f, 0.f, 0.f, 0.f, 0.f, 0.f};
    float pal1[8] = {0.f, 0.f, 0.f, 0.f, 0.f, 0.f, 0.f, 0.f};

    for (int nt = 0; nt < 16; ++nt) {
        const int n = nt * 16 + row;           // lane's output column
        const bf16_t* wbase = &Wl[(size_t)n * IN_DIM + hi * 16];

        v8f acc0 = {0.f, 0.f, 0.f, 0.f, 0.f, 0.f, 0.f, 0.f};
        v8f acc1 = {0.f, 0.f, 0.f, 0.f, 0.f, 0.f, 0.f, 0.f};
#pragma unroll
        for (int kt = 0; kt < 8; ++kt) {
            v16bf b = *reinterpret_cast<const v16bf*>(wbase + kt * 32);
            acc0 = __builtin_amdgcn_wmma_f32_16x16x32_bf16(
                false, a0[kt], false, b, (short)0, acc0, false, false);
            acc1 = __builtin_amdgcn_wmma_f32_16x16x32_bf16(
                false, a1[kt], false, b, (short)0, acc1, false, false);
        }
        // Pipeline: prefetch 2 K-steps of B frags, then alternate WMMA/DS.
        __builtin_amdgcn_sched_group_barrier(0x100, 4, 0);  // DS read x4
#pragma unroll
        for (int kt = 0; kt < 6; ++kt) {
            __builtin_amdgcn_sched_group_barrier(0x008, 2, 0);  // WMMA x2
            __builtin_amdgcn_sched_group_barrier(0x100, 2, 0);  // DS read x2
        }
        __builtin_amdgcn_sched_group_barrier(0x008, 4, 0);      // WMMA x4

        const float bias = abL[n];
        const float am   = amL[n];
#pragma unroll
        for (int r = 0; r < 8; ++r) {          // D: lane holds M = hi*8 + r
            pal0[r] += am * tanhf(acc0[r] + bias);
            pal1[r] += am * tanhf(acc1[r] + bias);
        }
    }

#pragma unroll
    for (int r = 0; r < 8; ++r) {              // reduce over the 16 N-lanes
        float v0 = pal0[r], v1 = pal1[r];
        v0 += __shfl_xor(v0, 1, 16);
        v0 += __shfl_xor(v0, 2, 16);
        v0 += __shfl_xor(v0, 4, 16);
        v0 += __shfl_xor(v0, 8, 16);
        v1 += __shfl_xor(v1, 1, 16);
        v1 += __shfl_xor(v1, 2, 16);
        v1 += __shfl_xor(v1, 4, 16);
        v1 += __shfl_xor(v1, 8, 16);
        if (row == 0) {
            alpha[group * 32 + hi * 8 + r]      = v0;
            alpha[group * 32 + 16 + hi * 8 + r] = v1;
        }
    }
}

// ---------------------------------------------------------------------------
// Kernel 2: segment max over dst (float max via ordered-uint atomicMax)
// ---------------------------------------------------------------------------
__global__ void k_segmax(const float* __restrict__ alpha,
                         const int* __restrict__ dst,
                         unsigned* __restrict__ amax)
{
    int e = blockIdx.x * blockDim.x + threadIdx.x;
    if (e < NE) atomicMax(&amax[dst[e]], ordU(alpha[e]));
}

// ---------------------------------------------------------------------------
// Kernel 3: a = exp(alpha - amax[dst]); denom[dst] += a
// ---------------------------------------------------------------------------
__global__ void k_expsum(const float* __restrict__ alpha,
                         const int* __restrict__ dst,
                         const unsigned* __restrict__ amax,
                         float* __restrict__ aexp, float* __restrict__ denom)
{
    int e = blockIdx.x * blockDim.x + threadIdx.x;
    if (e < NE) {
        float m = unordU(amax[dst[e]]);
        float v = expf(alpha[e] - m);
        aexp[e] = v;
        atomicAdd(&denom[dst[e]], v);
    }
}

// ---------------------------------------------------------------------------
// Kernel 4: agg[dst] += (a/denom) * x_j   (4 floats per thread)
// ---------------------------------------------------------------------------
__global__ void k_scatter(const float* __restrict__ x,
                          const int* __restrict__ src,
                          const int* __restrict__ dst,
                          const float* __restrict__ aexp,
                          const float* __restrict__ denom,
                          float* __restrict__ agg)
{
    long long idx = (long long)blockIdx.x * 256 + threadIdx.x;  // NE*64
    int e = (int)(idx >> 6);
    int c = (int)(idx & 63) * 4;
    int d = dst[e];
    float w = aexp[e] / (denom[d] + 1e-16f);
    const float4 v = *reinterpret_cast<const float4*>(
        x + (size_t)src[e] * IN_DIM + c);
    float* ag = agg + (size_t)d * IN_DIM + c;
    atomicAdd(ag + 0, w * v.x);
    atomicAdd(ag + 1, w * v.y);
    atomicAdd(ag + 2, w * v.z);
    atomicAdd(ag + 3, w * v.w);
}

// ---------------------------------------------------------------------------
// Kernel 5: out = agg@pwa^T + pwa_b + x@pwo^T + pwo_b (WMMA), + BN stats
// ---------------------------------------------------------------------------
__global__ __launch_bounds__(256) void k_node_out(
    const float* __restrict__ agg, const float* __restrict__ xf,
    const float* __restrict__ pwa, const float* __restrict__ pwo,
    const float* __restrict__ pwa_b, const float* __restrict__ pwo_b,
    float* __restrict__ outp, float* __restrict__ bsum,
    float* __restrict__ bsumsq)
{
    const int wave  = threadIdx.x >> 5;
    const int lane  = threadIdx.x & 31;
    const int group = blockIdx.x * 8 + wave;   // 16 nodes per group
    const int row   = lane & 15;
    const int hi    = lane >> 4;

    const size_t node = (size_t)(group * 16 + row);
    const float* __restrict__ ra = agg + node * IN_DIM;
    const float* __restrict__ rx = xf  + node * IN_DIM;

    v16bf aA[8], aX[8];
#pragma unroll
    for (int kt = 0; kt < 8; ++kt) {
        const int kb0 = kt * 32 + hi * 16;
        const int kb1 = kb0 + 8;
#pragma unroll
        for (int j = 0; j < 8; ++j) {
            aA[kt][j]     = (bf16_t)ra[kb0 + j];
            aA[kt][8 + j] = (bf16_t)ra[kb1 + j];
            aX[kt][j]     = (bf16_t)rx[kb0 + j];
            aX[kt][8 + j] = (bf16_t)rx[kb1 + j];
        }
    }

    for (int nt = 0; nt < 16; ++nt) {
        const int n = nt * 16 + row;
        const float* wa = pwa + (size_t)n * IN_DIM + hi * 16;
        const float* wo = pwo + (size_t)n * IN_DIM + hi * 16;

        v8f acc0 = {0.f, 0.f, 0.f, 0.f, 0.f, 0.f, 0.f, 0.f};
        v8f acc1 = {0.f, 0.f, 0.f, 0.f, 0.f, 0.f, 0.f, 0.f};
#pragma unroll
        for (int kt = 0; kt < 8; ++kt) {
            v16bf bA, bO;
#pragma unroll
            for (int j = 0; j < 16; ++j) {
                bA[j] = (bf16_t)wa[kt * 32 + j];
                bO[j] = (bf16_t)wo[kt * 32 + j];
            }
            acc0 = __builtin_amdgcn_wmma_f32_16x16x32_bf16(
                false, aA[kt], false, bA, (short)0, acc0, false, false);
            acc1 = __builtin_amdgcn_wmma_f32_16x16x32_bf16(
                false, aX[kt], false, bO, (short)0, acc1, false, false);
        }

        const float bias = pwa_b[n] + pwo_b[n];
        float s = 0.f, s2 = 0.f;
#pragma unroll
        for (int r = 0; r < 8; ++r) {
            float o = acc0[r] + acc1[r] + bias;
            outp[(size_t)(group * 16 + hi * 8 + r) * OUT_DIM + n] = o;
            s += o; s2 += o * o;
        }
        atomicAdd(&bsum[n], s);
        atomicAdd(&bsumsq[n], s2);
    }
}

// ---------------------------------------------------------------------------
// Kernel 6: BN stats finalize
// ---------------------------------------------------------------------------
__global__ void k_bnstats(const float* __restrict__ bsum,
                          const float* __restrict__ bsumsq,
                          float* __restrict__ mean, float* __restrict__ rstd)
{
    int n = threadIdx.x;
    float m = bsum[n] * (1.f / NTOT);
    float v = bsumsq[n] * (1.f / NTOT) - m * m;
    mean[n] = m;
    rstd[n] = rsqrtf(v + 1e-5f);
}

// ---------------------------------------------------------------------------
// Kernel 7: BN + SELU in place, fused sigmoid pooling score (wave per node)
// ---------------------------------------------------------------------------
__global__ __launch_bounds__(256) void k_bn_selu_score(
    float* __restrict__ outp, const float* __restrict__ mean,
    const float* __restrict__ rstd, const float* __restrict__ gamma,
    const float* __restrict__ beta, const float* __restrict__ yw,
    const float* __restrict__ ybp, float* __restrict__ y)
{
    const int wave = threadIdx.x >> 5;
    const int lane = threadIdx.x & 31;
    const int node = blockIdx.x * 8 + wave;
    float* rowp = outp + (size_t)node * OUT_DIM;
    const float yb = ybp[0];

    float dot = 0.f;
#pragma unroll
    for (int j = 0; j < 8; ++j) {
        int c = lane + 32 * j;
        float o = rowp[c];
        o = (o - mean[c]) * rstd[c] * gamma[c] + beta[c];
        o = (o > 0.f) ? SELU_SCALE * o
                      : SELU_SCALE * SELU_ALPHA * (expf(o) - 1.f);
        rowp[c] = o;
        dot += o * yw[c];
    }
    dot += __shfl_xor(dot, 16, 32);
    dot += __shfl_xor(dot,  8, 32);
    dot += __shfl_xor(dot,  4, 32);
    dot += __shfl_xor(dot,  2, 32);
    dot += __shfl_xor(dot,  1, 32);
    if (lane == 0) y[node] = 1.f / (1.f + expf(-(dot + yb)));
}

// ---------------------------------------------------------------------------
// Kernel 8: per-batch descending bitonic sort of 1024 (score,idx) pairs
// ---------------------------------------------------------------------------
__global__ __launch_bounds__(1024) void k_topk(const float* __restrict__ y,
                                               int* __restrict__ ids,
                                               float* __restrict__ vals)
{
    __shared__ float sv[NNODE];
    __shared__ int   si[NNODE];
    const int b = blockIdx.x, t = threadIdx.x;
    sv[t] = y[b * NNODE + t];
    si[t] = t;
    __syncthreads();

    for (int k = 2; k <= NNODE; k <<= 1) {
        for (int j = k >> 1; j > 0; j >>= 1) {
            int ixj = t ^ j;
            if (ixj > t) {
                bool desc = ((t & k) == 0);
                float a0 = sv[t], a1 = sv[ixj];
                bool sw = desc ? (a0 < a1) : (a0 > a1);
                if (sw) {
                    sv[t] = a1; sv[ixj] = a0;
                    int tmp = si[t]; si[t] = si[ixj]; si[ixj] = tmp;
                }
            }
            __syncthreads();
        }
    }
    if (t < KTOP) {
        ids[b * KTOP + t]  = si[t];
        vals[b * KTOP + t] = sv[t];
    }
}

// ---------------------------------------------------------------------------
// Kernel 9: gather top-k rows, gate by score
// ---------------------------------------------------------------------------
__global__ void k_gather(const float* __restrict__ outp,
                         const int* __restrict__ ids,
                         const float* __restrict__ vals,
                         float* __restrict__ out)
{
    int idx = blockIdx.x * 256 + threadIdx.x;   // BB*KTOP*OUT_DIM
    int d   = idx & (OUT_DIM - 1);
    int bj  = idx >> 8;
    int b   = bj / KTOP;
    int j   = bj - b * KTOP;
    int node = b * NNODE + ids[b * KTOP + j];
    out[idx] = outp[(size_t)node * OUT_DIM + d] * vals[b * KTOP + j];
}

// ---------------------------------------------------------------------------
extern "C" void kernel_launch(void* const* d_in, const int* in_sizes, int n_in,
                              void* d_out, int out_size, void* d_ws,
                              size_t ws_size, hipStream_t stream)
{
    const float* x       = (const float*)d_in[0];
    const int*   edges   = (const int*)d_in[1];
    const int*   src     = edges;            // edges[0] = source
    const int*   dst     = edges + NE;       // edges[1] = destination
    const float* att_w   = (const float*)d_in[2];
    const float* att_b   = (const float*)d_in[3];
    const float* att_map = (const float*)d_in[4];
    const float* pwa_w   = (const float*)d_in[5];
    const float* pwa_b   = (const float*)d_in[6];
    const float* pwo_w   = (const float*)d_in[7];
    const float* pwo_b   = (const float*)d_in[8];
    const float* gamma   = (const float*)d_in[9];
    const float* beta    = (const float*)d_in[10];
    const float* yw      = (const float*)d_in[11];
    const float* yb      = (const float*)d_in[12];

    // Workspace layout
    float*    alpha  = (float*)d_ws;                       // NE
    float*    aexp   = alpha + NE;                         // NE
    unsigned* amax   = (unsigned*)(aexp + NE);             // NTOT
    float*    denom  = (float*)(amax + NTOT);              // NTOT
    float*    agg    = denom + NTOT;                       // NTOT*IN_DIM
    float*    outp   = agg + (size_t)NTOT * IN_DIM;        // NTOT*OUT_DIM
    float*    bsum   = outp + (size_t)NTOT * OUT_DIM;      // OUT_DIM
    float*    bsumsq = bsum + OUT_DIM;                     // OUT_DIM
    float*    mean   = bsumsq + OUT_DIM;                   // OUT_DIM
    float*    rstd   = mean + OUT_DIM;                     // OUT_DIM
    float*    ysc    = rstd + OUT_DIM;                     // NTOT
    int*      ids    = (int*)(ysc + NTOT);                 // BB*KTOP
    float*    vals   = (float*)(ids + BB * KTOP);          // BB*KTOP

    hipMemsetAsync(amax, 0, NTOT * sizeof(unsigned), stream);
    hipMemsetAsync(denom, 0, NTOT * sizeof(float), stream);
    hipMemsetAsync(agg, 0, (size_t)NTOT * IN_DIM * sizeof(float), stream);
    hipMemsetAsync(bsum, 0, 2 * OUT_DIM * sizeof(float), stream);

    k_edge_alpha<<<NE / 32 / 8, 256, 0, stream>>>(x, src, dst, att_w, att_b,
                                                  att_map, alpha);
    k_segmax<<<NE / 256, 256, 0, stream>>>(alpha, dst, amax);
    k_expsum<<<NE / 256, 256, 0, stream>>>(alpha, dst, amax, aexp, denom);
    k_scatter<<<NE * 64 / 256, 256, 0, stream>>>(x, src, dst, aexp, denom, agg);
    k_node_out<<<NTOT / 16 / 8, 256, 0, stream>>>(agg, x, pwa_w, pwo_w, pwa_b,
                                                  pwo_b, outp, bsum, bsumsq);
    k_bnstats<<<1, OUT_DIM, 0, stream>>>(bsum, bsumsq, mean, rstd);
    k_bn_selu_score<<<NTOT / 8, 256, 0, stream>>>(outp, mean, rstd, gamma,
                                                  beta, yw, yb, ysc);
    k_topk<<<BB, 1024, 0, stream>>>(ysc, ids, vals);
    k_gather<<<(BB * KTOP * OUT_DIM) / 256, 256, 0, stream>>>(outp, ids, vals,
                                                              (float*)d_out);
}
